// GAT_RA_11501922419027
// MI455X (gfx1250) — matrix operand order repaired and checked by reading, not compiled, not run on hardware
//
#include <hip/hip_runtime.h>
#include <hip/hip_bf16.h>
#include <math.h>

// ---------------- problem constants (from reference) ----------------
#define N_NODES 50000
#define E_EDGES 800000
#define DIN     128
#define HID     32
#define HEADS   8
#define HD      (HEADS * HID)   // 256
#define DOUT    16
#define NEG_SLOPE 0.2f
#define BN_EPS    1e-5f

typedef __attribute__((ext_vector_type(16))) __bf16 v16bf;
typedef __attribute__((ext_vector_type(8)))  float  v8f;

// ---------------------------------------------------------------
// WMMA GEMM: C[M,N] = A[M,K] * B[K,N], f32 in/out, bf16 multiply,
// f32 accumulate (v_wmma_f32_16x16x32_bf16).
//   - one wave computes a 16 x (16*NT) strip of C (NT accumulators,
//     one A fragment reused NT times per K-step)
//   - 8 waves per block stacked along M
//   - K and N are compile-time -> K-loop fully unrolls and all
//     strided element addresses fold into the 24-bit IOFFSET field
//   - f32->bf16 via native packed converts (v_cvt_pk_bf16_f32)
// Requires: K % 32 == 0, M % 16 == 0, N % (16*NT) == 0.
// ---------------------------------------------------------------
template <int K, int NT, int N>
__global__ void wmma_gemm_bf16(const float* __restrict__ A,
                               const float* __restrict__ B,
                               float* __restrict__ C,
                               int M) {
  const int lane = threadIdx.x & 31;
  const int wave = threadIdx.x >> 5;
  const int mBase = (blockIdx.x * 8 + wave) * 16;
  if (mBase >= M) return;                 // whole wave exits uniformly
  const int nBase = blockIdx.y * (16 * NT);

  const int half = lane >> 4;             // 0: lanes 0-15, 1: lanes 16-31
  const int mRow = mBase + (lane & 15);   // A-fragment row  (M = lane%16)
  const int nCol = nBase + (lane & 15);   // B/C-fragment col (N = lane%16)

  v8f acc[NT];
#pragma unroll
  for (int t = 0; t < NT; ++t) acc[t] = (v8f){};

#pragma unroll
  for (int k0 = 0; k0 < K; k0 += 32) {
    // --- A fragment (16x32 bf16, ISA 7.12.2): lane half selects K octet.
    //     elem e in 0..7  -> K = k0 + half*8 + e
    //     elem e in 8..15 -> K = k0 + 16 + half*8 + (e-8)
    v16bf a;
    const float* Arow = A + (size_t)mRow * K + k0 + half * 8;
#pragma unroll
    for (int e = 0; e < 8; ++e) {
      a[e]     = (__bf16)Arow[e];
      a[e + 8] = (__bf16)Arow[16 + e];
    }
    // --- B fragments (32x16 bf16): N = lane%16, lane half selects K octet.
    //     elem e -> K = k0 + half*8 + e (e<8), k0 + 16 + half*8 + (e-8) (e>=8)
#pragma unroll
    for (int t = 0; t < NT; ++t) {
      v16bf b;
      const float* Bcol = B + (size_t)(k0 + half * 8) * N + nCol + t * 16;
#pragma unroll
      for (int e = 0; e < 16; ++e)
        b[e] = (__bf16)Bcol[(e + (e & 8)) * N];
      acc[t] = __builtin_amdgcn_wmma_f32_16x16x32_bf16(
          /*neg_a=*/false, a, /*neg_b=*/false, b,
          /*c_mod=*/(short)0, acc[t], /*reuse_a=*/false, /*reuse_b=*/false);
    }
  }
  // --- C store (32-bit 16x16 C/D layout): VGPR r -> M = mBase + r + half*8
#pragma unroll
  for (int t = 0; t < NT; ++t) {
    float* Crow = C + (size_t)(mBase + half * 8) * N + nCol + t * 16;
#pragma unroll
    for (int r = 0; r < 8; ++r)
      Crow[r * N] = acc[t][r];            // r*N folds into IOFFSET
  }
}

// ---------------------------------------------------------------
// per-node attention scores: s[n,h] = dot(xw[n,h,:], a_src[h,:]),
//                            d[n,h] = dot(xw[n,h,:], a_dst[h,:])
// ---------------------------------------------------------------
__global__ void attn_scores(const float* __restrict__ xw,
                            const float* __restrict__ a_src,
                            const float* __restrict__ a_dst,
                            float* __restrict__ s, float* __restrict__ d,
                            int n_nodes, int heads, int hid) {
  int i = blockIdx.x * blockDim.x + threadIdx.x;
  if (i >= n_nodes * heads) return;
  int h = i % heads;
  const float* row = xw + (size_t)(i / heads) * heads * hid + h * hid;
  const float* as = a_src + h * hid;
  const float* ad = a_dst + h * hid;
  float ss = 0.f, dd = 0.f;
  for (int f = 0; f < hid; ++f) { ss += row[f] * as[f]; dd += row[f] * ad[f]; }
  s[i] = ss; d[i] = dd;
}

__global__ void fill_f32(float* __restrict__ p, float v, int n) {
  int i = blockIdx.x * blockDim.x + threadIdx.x;
  if (i < n) p[i] = v;
}

__global__ void init_bias(float* __restrict__ out, const float* __restrict__ b,
                          int n_nodes, int c) {
  int i = blockIdx.x * blockDim.x + threadIdx.x;
  if (i < n_nodes * c) out[i] = b[i % c];
}

static __device__ __forceinline__ void edge_pair(const int* __restrict__ es,
                                                 const int* __restrict__ ed,
                                                 int e, int& u, int& v) {
  if (e < E_EDGES) { u = es[e]; v = ed[e]; }
  else             { u = v = e - E_EDGES; }   // appended self-loop
}

// segment max of leaky_relu(s[src]+d[dst]) per (dst, head)
__global__ void alpha_max(const int* __restrict__ es, const int* __restrict__ ed,
                          const float* __restrict__ s, const float* __restrict__ d,
                          float* __restrict__ amax, int etot, int heads) {
  int i = blockIdx.x * blockDim.x + threadIdx.x;
  if (i >= etot * heads) return;
  int e = i / heads, h = i - e * heads;
  int u, v; edge_pair(es, ed, e, u, v);
  float a = s[u * heads + h] + d[v * heads + h];
  a = a > 0.f ? a : NEG_SLOPE * a;
  atomicMax(&amax[v * heads + h], a);
}

// denominator: den[dst,h] += exp(alpha - amax[dst,h])
__global__ void softmax_den(const int* __restrict__ es, const int* __restrict__ ed,
                            const float* __restrict__ s, const float* __restrict__ d,
                            const float* __restrict__ amax, float* __restrict__ den,
                            int etot, int heads) {
  int i = blockIdx.x * blockDim.x + threadIdx.x;
  if (i >= etot * heads) return;
  int e = i / heads, h = i - e * heads;
  int u, v; edge_pair(es, ed, e, u, v);
  float a = s[u * heads + h] + d[v * heads + h];
  a = a > 0.f ? a : NEG_SLOPE * a;
  atomicAdd(&den[v * heads + h], __expf(a - amax[v * heads + h]));
}

// one block per edge; thread t = head*HID_T + feat; coefs via LDS broadcast
template <int HEADS_T, int HID_T>
__global__ void aggregate(const int* __restrict__ es, const int* __restrict__ ed,
                          const float* __restrict__ xw,
                          const float* __restrict__ s, const float* __restrict__ d,
                          const float* __restrict__ amax, const float* __restrict__ den,
                          float* __restrict__ out) {
  __shared__ float coef[HEADS_T];
  const int e = blockIdx.x;
  const int t = threadIdx.x;                 // 0 .. HEADS_T*HID_T-1
  int u, v; edge_pair(es, ed, e, u, v);
  if (t < HEADS_T) {
    float a = s[u * HEADS_T + t] + d[v * HEADS_T + t];
    a = a > 0.f ? a : NEG_SLOPE * a;
    coef[t] = __expf(a - amax[v * HEADS_T + t]) / den[v * HEADS_T + t];
  }
  __syncthreads();
  const float val = coef[t / HID_T] * xw[(size_t)u * (HEADS_T * HID_T) + t];
  atomicAdd(&out[(size_t)v * (HEADS_T * HID_T) + t], val);
}

__global__ void bn_elu(float* __restrict__ h, const float* __restrict__ g,
                       const float* __restrict__ be, const float* __restrict__ m,
                       const float* __restrict__ v, int n_nodes, int c) {
  int i = blockIdx.x * blockDim.x + threadIdx.x;
  if (i >= n_nodes * c) return;
  int ch = i % c;
  float x = (h[i] - m[ch]) * rsqrtf(v[ch] + BN_EPS) * g[ch] + be[ch];
  h[i] = x > 0.f ? x : (__expf(x) - 1.f);
}

__global__ void sigmoid_k(float* __restrict__ h, int n) {
  int i = blockIdx.x * blockDim.x + threadIdx.x;
  if (i < n) h[i] = 1.f / (1.f + __expf(-h[i]));
}

// ---------------------------------------------------------------
extern "C" void kernel_launch(void* const* d_in, const int* in_sizes, int n_in,
                              void* d_out, int out_size, void* d_ws, size_t ws_size,
                              hipStream_t stream) {
  (void)in_sizes; (void)n_in; (void)out_size; (void)ws_size;
  const float* x   = (const float*)d_in[0];
  const int*   es  = (const int*)d_in[1];
  const int*   ed  = (const int*)d_in[2];
  const float* W1  = (const float*)d_in[3];
  const float* as1 = (const float*)d_in[4];
  const float* ad1 = (const float*)d_in[5];
  const float* b1  = (const float*)d_in[6];
  const float* g1  = (const float*)d_in[7];
  const float* be1 = (const float*)d_in[8];
  const float* m1  = (const float*)d_in[9];
  const float* v1  = (const float*)d_in[10];
  const float* W2  = (const float*)d_in[11];
  const float* as2 = (const float*)d_in[12];
  const float* ad2 = (const float*)d_in[13];
  const float* b2  = (const float*)d_in[14];
  const float* g2  = (const float*)d_in[15];
  const float* be2 = (const float*)d_in[16];
  const float* m2  = (const float*)d_in[17];
  const float* v2  = (const float*)d_in[18];
  const float* W3  = (const float*)d_in[19];
  const float* as3 = (const float*)d_in[20];
  const float* ad3 = (const float*)d_in[21];
  const float* b3  = (const float*)d_in[22];

  // workspace layout (floats)
  float* ws   = (float*)d_ws;
  float* xw   = ws;                              // N*HD   (also reused for xw3)
  float* hbuf = xw + (size_t)N_NODES * HD;       // N*HD
  float* sbuf = hbuf + (size_t)N_NODES * HD;     // N*HEADS (layer3 reuses: N)
  float* dbuf = sbuf + (size_t)N_NODES * HEADS;
  float* amax = dbuf + (size_t)N_NODES * HEADS;
  float* den  = amax + (size_t)N_NODES * HEADS;

  const int ETOT = E_EDGES + N_NODES;
  const int TPB = 256;
  const int mBlocks = ((N_NODES / 16) + 7) / 8;  // 8 waves (16-row tiles) per block
  auto cdiv = [](int a, int b) { return (a + b - 1) / b; };

  // ======================= layer 1 =======================
  wmma_gemm_bf16<DIN, 4, HD><<<dim3(mBlocks, HD / 64), TPB, 0, stream>>>(x, W1, xw, N_NODES);
  attn_scores<<<cdiv(N_NODES * HEADS, TPB), TPB, 0, stream>>>(xw, as1, ad1, sbuf, dbuf, N_NODES, HEADS, HID);
  fill_f32<<<cdiv(N_NODES * HEADS, TPB), TPB, 0, stream>>>(amax, -1e30f, N_NODES * HEADS);
  fill_f32<<<cdiv(N_NODES * HEADS, TPB), TPB, 0, stream>>>(den, 0.f, N_NODES * HEADS);
  alpha_max<<<cdiv(ETOT * HEADS, TPB), TPB, 0, stream>>>(es, ed, sbuf, dbuf, amax, ETOT, HEADS);
  softmax_den<<<cdiv(ETOT * HEADS, TPB), TPB, 0, stream>>>(es, ed, sbuf, dbuf, amax, den, ETOT, HEADS);
  init_bias<<<cdiv(N_NODES * HD, TPB), TPB, 0, stream>>>(hbuf, b1, N_NODES, HD);
  aggregate<HEADS, HID><<<ETOT, HEADS * HID, 0, stream>>>(es, ed, xw, sbuf, dbuf, amax, den, hbuf);
  bn_elu<<<cdiv(N_NODES * HD, TPB), TPB, 0, stream>>>(hbuf, g1, be1, m1, v1, N_NODES, HD);

  // ======================= layer 2 =======================
  wmma_gemm_bf16<HD, 4, HD><<<dim3(mBlocks, HD / 64), TPB, 0, stream>>>(hbuf, W2, xw, N_NODES);
  attn_scores<<<cdiv(N_NODES * HEADS, TPB), TPB, 0, stream>>>(xw, as2, ad2, sbuf, dbuf, N_NODES, HEADS, HID);
  fill_f32<<<cdiv(N_NODES * HEADS, TPB), TPB, 0, stream>>>(amax, -1e30f, N_NODES * HEADS);
  fill_f32<<<cdiv(N_NODES * HEADS, TPB), TPB, 0, stream>>>(den, 0.f, N_NODES * HEADS);
  alpha_max<<<cdiv(ETOT * HEADS, TPB), TPB, 0, stream>>>(es, ed, sbuf, dbuf, amax, ETOT, HEADS);
  softmax_den<<<cdiv(ETOT * HEADS, TPB), TPB, 0, stream>>>(es, ed, sbuf, dbuf, amax, den, ETOT, HEADS);
  // GEMM above already consumed hbuf -> safe to overwrite as layer-2 output
  init_bias<<<cdiv(N_NODES * HD, TPB), TPB, 0, stream>>>(hbuf, b2, N_NODES, HD);
  aggregate<HEADS, HID><<<ETOT, HEADS * HID, 0, stream>>>(es, ed, xw, sbuf, dbuf, amax, den, hbuf);
  bn_elu<<<cdiv(N_NODES * HD, TPB), TPB, 0, stream>>>(hbuf, g2, be2, m2, v2, N_NODES, HD);

  // ======================= layer 3 (1 head, DOUT=16, mean==identity) =========
  float* xw3 = xw;          // overlay: only N*DOUT needed
  wmma_gemm_bf16<HD, 1, DOUT><<<dim3(mBlocks, DOUT / 16), TPB, 0, stream>>>(hbuf, W3, xw3, N_NODES);
  attn_scores<<<cdiv(N_NODES, TPB), TPB, 0, stream>>>(xw3, as3, ad3, sbuf, dbuf, N_NODES, 1, DOUT);
  fill_f32<<<cdiv(N_NODES, TPB), TPB, 0, stream>>>(amax, -1e30f, N_NODES);
  fill_f32<<<cdiv(N_NODES, TPB), TPB, 0, stream>>>(den, 0.f, N_NODES);
  alpha_max<<<cdiv(ETOT, TPB), TPB, 0, stream>>>(es, ed, sbuf, dbuf, amax, ETOT, 1);
  softmax_den<<<cdiv(ETOT, TPB), TPB, 0, stream>>>(es, ed, sbuf, dbuf, amax, den, ETOT, 1);
  float* out = (float*)d_out;
  init_bias<<<cdiv(N_NODES * DOUT, TPB), TPB, 0, stream>>>(out, b3, N_NODES, DOUT);
  aggregate<1, DOUT><<<ETOT, DOUT, 0, stream>>>(es, ed, xw3, sbuf, dbuf, amax, den, out);
  sigmoid_k<<<cdiv(N_NODES * DOUT, TPB), TPB, 0, stream>>>(out, N_NODES * DOUT);
}
